// ScaledIntegratedReasoning_47588237640064
// MI455X (gfx1250) — compile-verified, matrix operand order checked by reading, not run on hardware
//
#include <hip/hip_runtime.h>
#include <hip/hip_bf16.h>
#include <math.h>

typedef __bf16 bf16_t;
typedef __bf16 v8bf  __attribute__((ext_vector_type(8)));
typedef __bf16 v16bf __attribute__((ext_vector_type(16)));
typedef float  v8f   __attribute__((ext_vector_type(8)));

#define B_SZ   4
#define S_LEN  2048
#define H_DIM  1024
#define R_DIM  512
#define NHEAD  8
#define HDH    64
#define M_TOT  (B_SZ * S_LEN)   // 8192

// ---------------------------------------------------------------------------
// bf16 fragment assembly: two 16B LDS loads -> one 32B v16bf A/B fragment.
// Layout per CDNA5 ISA 7.12.2 (16-bit A 16x32): lanes 0-15 hold K{0..7,16..23},
// lanes 16-31 hold K{8..15,24..31}; B mirrors with N across lanes.
// ---------------------------------------------------------------------------
__device__ inline v16bf frag16(const bf16_t* base /*LDS*/, int off) {
    v8bf lo = *(const v8bf*)(base + off);
    v8bf hi = *(const v8bf*)(base + off + 16);
    return __builtin_shufflevector(lo, hi, 0,1,2,3,4,5,6,7,8,9,10,11,12,13,14,15);
}

// ---------------------------------------------------------------------------
// Generic bf16 WMMA GEMM: C[M,N] = A[M,K](bf16,row-major,lda) @ Wt[N,K]^T + bias
// Block tile 128x128, 8 waves as 2(M) x 4(N), wave tile 64x32, K-step 32,
// double-buffered LDS staging (one barrier per K-step).
// Compile-time epilogue: F32OUT ? fp32 store : bf16 store; optional RELU.
// ---------------------------------------------------------------------------
template<bool RELU, bool F32OUT>
__global__ __launch_bounds__(256, 2) void wmma_gemm_kernel(
    const bf16_t* __restrict__ A, int lda,
    const bf16_t* __restrict__ Wt, int K,
    const float* __restrict__ bias,
    void* __restrict__ Cout, int Ntot)
{
    __shared__ __align__(16) bf16_t As[2][128 * 32];
    __shared__ __align__(16) bf16_t Ws[2][128 * 32];

    const int t    = threadIdx.x;
    const int lane = t & 31;
    const int wave = t >> 5;
    const int half = lane >> 4;
    const int l15  = lane & 15;
    const int mw   = (wave & 1) * 64;   // wave M offset in block tile
    const int nw   = (wave >> 1) * 32;  // wave N offset in block tile
    const int m0   = blockIdx.y * 128;
    const int n0   = blockIdx.x * 128;

    // staging indices (constant across K loop)
    const int lin0 = t * 2;
    const int srow = lin0 >> 2;           // 0..127
    const int sseg = (lin0 & 3) * 8;      // 0 or 16 (i adds 8)

    auto stage = [&](int buf, int k0) {
        for (int i = 0; i < 2; i++) {
            int row = srow;
            int seg = sseg + i * 8;
            *(v8bf*)&As[buf][row * 32 + seg] =
                *(const v8bf*)(A + (size_t)(m0 + row) * lda + k0 + seg);
            *(v8bf*)&Ws[buf][row * 32 + seg] =
                *(const v8bf*)(Wt + (size_t)(n0 + row) * K + k0 + seg);
        }
    };

    v8f zero8 = {0.f,0.f,0.f,0.f,0.f,0.f,0.f,0.f};
    v8f acc[4][2];
    for (int mi = 0; mi < 4; mi++)
        for (int ni = 0; ni < 2; ni++) acc[mi][ni] = zero8;

    stage(0, 0);
    __syncthreads();

    int buf = 0;
    const int koff = half * 8;
    for (int k0 = 0; k0 < K; k0 += 32) {
        if (k0 + 32 < K) stage(buf ^ 1, k0 + 32);   // prefetch next tile

        v16bf af[4], bfm[2];
        for (int mi = 0; mi < 4; mi++)
            af[mi] = frag16(&As[buf][0], (mw + mi * 16 + l15) * 32 + koff);
        for (int ni = 0; ni < 2; ni++)
            bfm[ni] = frag16(&Ws[buf][0], (nw + ni * 16 + l15) * 32 + koff);

        for (int mi = 0; mi < 4; mi++)
            for (int ni = 0; ni < 2; ni++)
                acc[mi][ni] = __builtin_amdgcn_wmma_f32_16x16x32_bf16(
                    false, af[mi], false, bfm[ni], (short)0, acc[mi][ni], false, false);

        __syncthreads();
        buf ^= 1;
    }

    // Epilogue: C layout = VGPR j, lanes 0-15 -> M=j, lanes 16-31 -> M=j+8.
    float* Cf  = (float*)Cout;
    bf16_t* Cb = (bf16_t*)Cout;
    for (int mi = 0; mi < 4; mi++)
        for (int ni = 0; ni < 2; ni++) {
            const int row0 = m0 + mw + mi * 16 + half * 8;
            const int col  = n0 + nw + ni * 16 + l15;
            const float bv = bias[col];
            const size_t base = (size_t)row0 * Ntot + col;
            for (int j = 0; j < 8; j++) {
                float v = acc[mi][ni][j] + bv;
                if (RELU) v = fmaxf(v, 0.f);
                if (F32OUT) Cf[base + (size_t)j * Ntot] = v;
                else        Cb[base + (size_t)j * Ntot] = (bf16_t)v;
            }
        }
}

// ---------------------------------------------------------------------------
// Flash attention: grid (S/128, NH, B), 256 threads. Q block 128 rows, K/V
// blocks of 64 rows streamed with online softmax. All matmuls via bf16 WMMA.
// Reads fp32 qkv [M, 3R]; writes bf16 attention output [M, R].
// ---------------------------------------------------------------------------
__global__ __launch_bounds__(256, 2) void flash_attn_kernel(
    const float* __restrict__ qkv, bf16_t* __restrict__ o)
{
    __shared__ __align__(16) bf16_t Qs[128 * 64];   // [qrow][dim]
    __shared__ __align__(16) bf16_t Ks[64 * 64];    // [krow][dim]  (B frag for QK^T)
    __shared__ __align__(16) bf16_t Vts[64 * 64];   // [dim][krow]  (B frag for P@V)
    __shared__ __align__(16) bf16_t Ps[8 * 16 * 64];// per-wave 16x64 P tile

    const int t = threadIdx.x, lane = t & 31, wave = t >> 5;
    const int half = lane >> 4, l15 = lane & 15;
    const int qb = blockIdx.x, h = blockIdx.y, b = blockIdx.z;
    const size_t qrow0 = (size_t)b * S_LEN + qb * 128;

    // Load + convert Q tile
    for (int i = 0; i < 32; i++) {
        int idx = t + i * 256;
        int r = idx >> 6, d = idx & 63;
        Qs[r * 64 + d] = (bf16_t)qkv[(qrow0 + r) * (3 * R_DIM) + h * HDH + d];
    }
    __syncthreads();

    v16bf qf[2];
    for (int ks = 0; ks < 2; ks++)
        qf[ks] = frag16(Qs, (wave * 16 + l15) * 64 + ks * 32 + half * 8);

    v8f zero8 = {0.f,0.f,0.f,0.f,0.f,0.f,0.f,0.f};
    v8f Oacc[4];
    for (int n = 0; n < 4; n++) Oacc[n] = zero8;
    float m[8], lsum[8];
    for (int j = 0; j < 8; j++) { m[j] = -1e30f; lsum[j] = 0.f; }

    for (int kb = 0; kb < S_LEN / 64; kb++) {
        __syncthreads();  // previous iteration done with Ks/Vts
        for (int i = 0; i < 16; i++) {
            int idx = t + i * 256;
            int r = idx >> 6, d = idx & 63;
            size_t g = ((size_t)b * S_LEN + kb * 64 + r) * (3 * R_DIM) + h * HDH + d;
            Ks[r * 64 + d]  = (bf16_t)qkv[g + R_DIM];       // K
            Vts[d * 64 + r] = (bf16_t)qkv[g + 2 * R_DIM];   // V transposed
        }
        __syncthreads();

        // S = (Q K^T) / sqrt(64)
        v8f Sacc[4];
        for (int n = 0; n < 4; n++) Sacc[n] = zero8;
        for (int ks = 0; ks < 2; ks++)
            for (int n = 0; n < 4; n++) {
                v16bf kf = frag16(Ks, (n * 16 + l15) * 64 + ks * 32 + half * 8);
                Sacc[n] = __builtin_amdgcn_wmma_f32_16x16x32_bf16(
                    false, qf[ks], false, kf, (short)0, Sacc[n], false, false);
            }

        // online softmax (row stats per VGPR-slot j, reduced over 16-lane group)
        float mn[8];
        for (int j = 0; j < 8; j++) mn[j] = m[j];
        for (int n = 0; n < 4; n++)
            for (int j = 0; j < 8; j++) {
                Sacc[n][j] *= 0.125f;
                mn[j] = fmaxf(mn[j], Sacc[n][j]);
            }
        for (int mask = 1; mask < 16; mask <<= 1)
            for (int j = 0; j < 8; j++)
                mn[j] = fmaxf(mn[j], __shfl_xor(mn[j], mask, 32));

        float alpha[8], rs[8];
        for (int j = 0; j < 8; j++) {
            alpha[j] = expf(m[j] - mn[j]);
            rs[j] = 0.f;
            m[j] = mn[j];
        }
        for (int n = 0; n < 4; n++)
            for (int j = 0; j < 8; j++) {
                float pv = expf(Sacc[n][j] - mn[j]);
                Sacc[n][j] = pv;
                rs[j] += pv;
            }
        for (int mask = 1; mask < 16; mask <<= 1)
            for (int j = 0; j < 8; j++)
                rs[j] += __shfl_xor(rs[j], mask, 32);
        for (int j = 0; j < 8; j++) lsum[j] = lsum[j] * alpha[j] + rs[j];
        for (int n = 0; n < 4; n++)
            for (int j = 0; j < 8; j++) Oacc[n][j] *= alpha[j];

        // spill P (C layout) to per-wave LDS, reload as A fragments
        for (int n = 0; n < 4; n++)
            for (int j = 0; j < 8; j++)
                Ps[wave * 1024 + (j + half * 8) * 64 + n * 16 + l15] = (bf16_t)Sacc[n][j];
        __syncthreads();

        // O += P @ V
        for (int ks = 0; ks < 2; ks++) {
            v16bf pf = frag16(Ps, wave * 1024 + l15 * 64 + ks * 32 + half * 8);
            for (int n = 0; n < 4; n++) {
                v16bf vf = frag16(Vts, (n * 16 + l15) * 64 + ks * 32 + half * 8);
                Oacc[n] = __builtin_amdgcn_wmma_f32_16x16x32_bf16(
                    false, pf, false, vf, (short)0, Oacc[n], false, false);
            }
        }
    }

    for (int j = 0; j < 8; j++) {
        float inv = 1.f / lsum[j];
        for (int n = 0; n < 4; n++) {
            size_t row = qrow0 + wave * 16 + j + half * 8;
            int col = h * HDH + n * 16 + l15;
            o[row * R_DIM + col] = (bf16_t)(Oacc[n][j] * inv);
        }
    }
}

// ---------------------------------------------------------------------------
// Row LayerNorm: one block per row; optional fp32 and bf16 (stride ldb) outs.
// ---------------------------------------------------------------------------
__global__ __launch_bounds__(256) void row_ln_kernel(
    const float* __restrict__ in, const float* __restrict__ g,
    const float* __restrict__ bb, float* __restrict__ outf,
    bf16_t* __restrict__ outb, int ldb, int N)
{
    int row = blockIdx.x, t = threadIdx.x;
    int cnt = N >> 8;  // 2 or 4
    float vloc[4];
    float s = 0.f, sq = 0.f;
    for (int i = 0; i < cnt; i++) {
        float v = in[(size_t)row * N + t + i * 256];
        vloc[i] = v; s += v; sq += v * v;
    }
    __shared__ float rs[256], rq[256];
    __shared__ float mean_s, rstd_s;
    rs[t] = s; rq[t] = sq;
    __syncthreads();
    for (int off = 128; off > 0; off >>= 1) {
        if (t < off) { rs[t] += rs[t + off]; rq[t] += rq[t + off]; }
        __syncthreads();
    }
    if (t == 0) {
        float mean = rs[0] / N;
        float var = rq[0] / N - mean * mean;
        mean_s = mean; rstd_s = rsqrtf(var + 1e-5f);
    }
    __syncthreads();
    float mean = mean_s, rstd = rstd_s;
    for (int i = 0; i < cnt; i++) {
        int c = t + i * 256;
        float v = (vloc[i] - mean) * rstd * g[c] + bb[c];
        if (outf) outf[(size_t)row * N + c] = v;
        if (outb) outb[(size_t)row * ldb + c] = (bf16_t)v;
    }
}

// res += add, then LayerNorm(res) -> bf16 (residual branch)
__global__ __launch_bounds__(256) void add_ln_kernel(
    float* __restrict__ res, const float* __restrict__ add,
    const float* __restrict__ g, const float* __restrict__ bb,
    bf16_t* __restrict__ outb, int N)
{
    int row = blockIdx.x, t = threadIdx.x;
    int cnt = N >> 8;
    float vloc[4];
    float s = 0.f, sq = 0.f;
    for (int i = 0; i < cnt; i++) {
        size_t idx = (size_t)row * N + t + i * 256;
        float v = res[idx] + add[idx];
        res[idx] = v;
        vloc[i] = v; s += v; sq += v * v;
    }
    __shared__ float rs[256], rq[256];
    __shared__ float mean_s, rstd_s;
    rs[t] = s; rq[t] = sq;
    __syncthreads();
    for (int off = 128; off > 0; off >>= 1) {
        if (t < off) { rs[t] += rs[t + off]; rq[t] += rq[t + off]; }
        __syncthreads();
    }
    if (t == 0) {
        float mean = rs[0] / N;
        float var = rq[0] / N - mean * mean;
        mean_s = mean; rstd_s = rsqrtf(var + 1e-5f);
    }
    __syncthreads();
    float mean = mean_s, rstd = rstd_s;
    for (int i = 0; i < cnt; i++) {
        int c = t + i * 256;
        outb[(size_t)row * N + c] = (bf16_t)((vloc[i] - mean) * rstd * g[c] + bb[c]);
    }
}

// outb = bf16(res + add); res updated
__global__ void add_bf16_kernel(float* __restrict__ res,
                                const float* __restrict__ add,
                                bf16_t* __restrict__ outb, size_t n)
{
    size_t i = (size_t)blockIdx.x * 256 + threadIdx.x;
    if (i < n) { float v = res[i] + add[i]; res[i] = v; outb[i] = (bf16_t)v; }
}

// out = hidden + sigmoid(g2) * reasoned
__global__ void final_gate_kernel(const float* __restrict__ hs,
                                  const float* __restrict__ g2,
                                  const float* __restrict__ rz,
                                  float* __restrict__ out, size_t n)
{
    size_t i = (size_t)blockIdx.x * 256 + threadIdx.x;
    if (i < n) {
        float s = 1.f / (1.f + expf(-g2[i]));
        out[i] = hs[i] + s * rz[i];
    }
}

// wT[n*K+k] = bf16(w[k*N+n])  (transpose + convert, run once per weight)
__global__ void conv_wT_kernel(const float* __restrict__ w,
                               bf16_t* __restrict__ wT, int K, int N)
{
    size_t i = (size_t)blockIdx.x * 256 + threadIdx.x;  // i = n*K + k
    int n = (int)(i / K);
    int k = (int)(i - (size_t)n * K);
    wT[i] = (bf16_t)w[(size_t)k * N + n];
}

// y[row*ld_out + col] = bf16(x[row*n_in + col])
__global__ void f2bf_strided_kernel(const float* __restrict__ x,
                                    bf16_t* __restrict__ y, int n_in, int ld_out)
{
    size_t i = (size_t)blockIdx.x * 256 + threadIdx.x;
    int row = (int)(i / n_in);
    int col = (int)(i - (size_t)row * n_in);
    y[(size_t)row * ld_out + col] = (bf16_t)x[i];
}

// pooled[b*H+h] = mean over S of hidden[b,:,h]
__global__ void pool_kernel(const float* __restrict__ hs, float* __restrict__ pooled)
{
    int i = blockIdx.x * 256 + threadIdx.x;  // b*H + h
    int b = i >> 10, h = i & 1023;
    const float* p = hs + (size_t)b * S_LEN * H_DIM + h;
    float s = 0.f;
    for (int ss = 0; ss < S_LEN; ss++) s += p[(size_t)ss * H_DIM];
    pooled[i] = s * (1.f / S_LEN);
}

// complexity assessor MLP (tiny; one block)
__global__ __launch_bounds__(256) void ca_kernel(
    const float* __restrict__ pooled,
    const float* __restrict__ w1, const float* __restrict__ b1,
    const float* __restrict__ w2, const float* __restrict__ b2,
    const float* __restrict__ w3, const float* __restrict__ b3,
    float* __restrict__ scores)
{
    __shared__ float h1[B_SZ][256];
    __shared__ float h2[B_SZ][32];
    int t = threadIdx.x;
    for (int b = 0; b < B_SZ; b++) {
        float s = b1[t];
        for (int k = 0; k < 1024; k++) s += pooled[b * 1024 + k] * w1[k * 256 + t];
        h1[b][t] = fmaxf(s, 0.f);
    }
    __syncthreads();
    if (t < 32)
        for (int b = 0; b < B_SZ; b++) {
            float s = b2[t];
            for (int k = 0; k < 256; k++) s += h1[b][k] * w2[k * 32 + t];
            h2[b][t] = fmaxf(s, 0.f);
        }
    __syncthreads();
    if (t < B_SZ) {
        float s = b3[0];
        for (int k = 0; k < 32; k++) s += h2[t][k] * w3[k];
        scores[t] = 1.f / (1.f + expf(-s));
    }
}

// ---------------------------------------------------------------------------
extern "C" void kernel_launch(void* const* d_in, const int* in_sizes, int n_in,
                              void* d_out, int out_size, void* d_ws, size_t ws_size,
                              hipStream_t stream)
{
    const float* hs    = (const float*)d_in[0];
    const float* ca_w1 = (const float*)d_in[1];
    const float* ca_b1 = (const float*)d_in[2];
    const float* ca_w2 = (const float*)d_in[3];
    const float* ca_b2 = (const float*)d_in[4];
    const float* ca_w3 = (const float*)d_in[5];
    const float* ca_b3 = (const float*)d_in[6];
    const float* rp_w  = (const float*)d_in[7];
    const float* rp_b  = (const float*)d_in[8];
    const float* rn_g  = (const float*)d_in[9];
    const float* rn_b  = (const float*)d_in[10];
    const float* ln1_g = (const float*)d_in[11];
    const float* ln1_b = (const float*)d_in[12];
    const float* wqkv  = (const float*)d_in[13];
    const float* bqkv  = (const float*)d_in[14];
    const float* wo    = (const float*)d_in[15];
    const float* bo    = (const float*)d_in[16];
    const float* ln2_g = (const float*)d_in[17];
    const float* ln2_b = (const float*)d_in[18];
    const float* ff1_w = (const float*)d_in[19];
    const float* ff1_b = (const float*)d_in[20];
    const float* ff2_w = (const float*)d_in[21];
    const float* ff2_b = (const float*)d_in[22];
    const float* op_w  = (const float*)d_in[23];
    const float* op_b  = (const float*)d_in[24];
    const float* on_g  = (const float*)d_in[25];
    const float* on_b  = (const float*)d_in[26];
    const float* gw1   = (const float*)d_in[27];
    const float* gb1   = (const float*)d_in[28];
    const float* gw2   = (const float*)d_in[29];
    const float* gb2   = (const float*)d_in[30];
    float* out = (float*)d_out;

    // bump allocator over d_ws
    char* p = (char*)d_ws;
    auto alloc = [&](size_t bytes) -> void* {
        void* r = (void*)p;
        p += (bytes + 255) & ~(size_t)255;
        return r;
    };
    const size_t M = M_TOT;
    bf16_t* gatein   = (bf16_t*)alloc(M * 2 * H_DIM * 2);      // [M, 2H] bf16
    bf16_t* rpT      = (bf16_t*)alloc((size_t)R_DIM * H_DIM * 2);
    bf16_t* qkvT     = (bf16_t*)alloc((size_t)3 * R_DIM * R_DIM * 2);
    bf16_t* woT      = (bf16_t*)alloc((size_t)R_DIM * R_DIM * 2);
    bf16_t* ff1T     = (bf16_t*)alloc((size_t)4 * R_DIM * R_DIM * 2);
    bf16_t* ff2T     = (bf16_t*)alloc((size_t)R_DIM * 4 * R_DIM * 2);
    bf16_t* opT      = (bf16_t*)alloc((size_t)H_DIM * R_DIM * 2);
    bf16_t* gw1T     = (bf16_t*)alloc((size_t)H_DIM * 2 * H_DIM * 2);
    bf16_t* gw2T     = (bf16_t*)alloc((size_t)H_DIM * H_DIM * 2);
    float*  pooled   = (float*) alloc((size_t)B_SZ * H_DIM * 4);
    float*  buf1     = (float*) alloc(M * 3 * R_DIM * 4);      // qkv / t3 / g2
    float*  tmp      = (float*) alloc(M * R_DIM * 4);          // pre-LN / branch
    float*  r_f32    = (float*) alloc(M * R_DIM * 4);          // residual stream
    bf16_t* y_bf16   = (bf16_t*)alloc(M * R_DIM * 2);          // LN outs (y1,y2)
    bf16_t* o_bf16   = (bf16_t*)alloc(M * R_DIM * 2);          // attn out / r_bf16
    bf16_t* h1_bf16  = (bf16_t*)alloc(M * 4 * R_DIM * 2);      // relu(ff1)
    float*  reasoned = (float*) alloc(M * H_DIM * 4);
    bf16_t* g1_bf16  = (bf16_t*)alloc(M * H_DIM * 2);

    auto gemm_f32 = [&](const bf16_t* A, int lda, const bf16_t* Wt, int K,
                        const float* bias, float* Cf, int N) {
        dim3 grid(N / 128, M_TOT / 128);
        wmma_gemm_kernel<false, true><<<grid, 256, 0, stream>>>(
            A, lda, Wt, K, bias, (void*)Cf, N);
    };
    auto gemm_relu_bf16 = [&](const bf16_t* A, int lda, const bf16_t* Wt, int K,
                              const float* bias, bf16_t* Cb, int N) {
        dim3 grid(N / 128, M_TOT / 128);
        wmma_gemm_kernel<true, false><<<grid, 256, 0, stream>>>(
            A, lda, Wt, K, bias, (void*)Cb, N);
    };

    // Weight convert+transpose (bf16 [N,K])
    conv_wT_kernel<<<(512u*1024)/256,  256, 0, stream>>>(rp_w,  rpT,  1024, 512);
    conv_wT_kernel<<<(1536u*512)/256,  256, 0, stream>>>(wqkv,  qkvT, 512, 1536);
    conv_wT_kernel<<<(512u*512)/256,   256, 0, stream>>>(wo,    woT,  512, 512);
    conv_wT_kernel<<<(2048u*512)/256,  256, 0, stream>>>(ff1_w, ff1T, 512, 2048);
    conv_wT_kernel<<<(512u*2048)/256,  256, 0, stream>>>(ff2_w, ff2T, 2048, 512);
    conv_wT_kernel<<<(1024u*512)/256,  256, 0, stream>>>(op_w,  opT,  512, 1024);
    conv_wT_kernel<<<(1024u*2048)/256, 256, 0, stream>>>(gw1,   gw1T, 2048, 1024);
    conv_wT_kernel<<<(1024u*1024)/256, 256, 0, stream>>>(gw2,   gw2T, 1024, 1024);

    // hidden -> bf16 into gatein[:, 0:H]
    f2bf_strided_kernel<<<(unsigned)(M * H_DIM / 256), 256, 0, stream>>>(
        hs, gatein, H_DIM, 2 * H_DIM);

    // complexity scores -> out[M*H ..]
    pool_kernel<<<(B_SZ * H_DIM) / 256, 256, 0, stream>>>(hs, pooled);
    ca_kernel<<<1, 256, 0, stream>>>(pooled, ca_w1, ca_b1, ca_w2, ca_b2,
                                     ca_w3, ca_b3, out + (size_t)M * H_DIM);

    // r = LN(hidden @ rp_w + rp_b)
    gemm_f32(gatein, 2 * H_DIM, rpT, H_DIM, rp_b, tmp, R_DIM);
    row_ln_kernel<<<M_TOT, 256, 0, stream>>>(tmp, rn_g, rn_b, r_f32, nullptr, R_DIM, R_DIM);

    // y1 = LN(r);  qkv = y1 @ wqkv + bqkv
    row_ln_kernel<<<M_TOT, 256, 0, stream>>>(r_f32, ln1_g, ln1_b, nullptr, y_bf16, R_DIM, R_DIM);
    gemm_f32(y_bf16, R_DIM, qkvT, R_DIM, bqkv, buf1, 3 * R_DIM);

    // flash attention -> o_bf16
    flash_attn_kernel<<<dim3(S_LEN / 128, NHEAD, B_SZ), 256, 0, stream>>>(buf1, o_bf16);

    // r += o @ wo + bo ; y2 = LN(r)
    gemm_f32(o_bf16, R_DIM, woT, R_DIM, bo, tmp, R_DIM);
    add_ln_kernel<<<M_TOT, 256, 0, stream>>>(r_f32, tmp, ln2_g, ln2_b, y_bf16, R_DIM);

    // r += relu(y2 @ ff1 + b) @ ff2 + b
    gemm_relu_bf16(y_bf16, R_DIM, ff1T, R_DIM, ff1_b, h1_bf16, 4 * R_DIM);
    gemm_f32(h1_bf16, 4 * R_DIM, ff2T, 4 * R_DIM, ff2_b, tmp, R_DIM);
    add_bf16_kernel<<<(unsigned)(M * R_DIM / 256), 256, 0, stream>>>(
        r_f32, tmp, o_bf16 /*reuse as r_bf16*/, M * R_DIM);

    // reasoned = LN(r @ op_w + op_b); bf16 copy into gatein[:, H:2H]
    gemm_f32(o_bf16, R_DIM, opT, R_DIM, op_b, buf1, H_DIM);
    row_ln_kernel<<<M_TOT, 256, 0, stream>>>(buf1, on_g, on_b, reasoned,
                                             gatein + H_DIM, 2 * H_DIM, H_DIM);

    // gate: g = sigmoid(relu(gatein @ gw1 + gb1) @ gw2 + gb2)
    gemm_relu_bf16(gatein, 2 * H_DIM, gw1T, 2 * H_DIM, gb1, g1_bf16, H_DIM);
    gemm_f32(g1_bf16, H_DIM, gw2T, H_DIM, gb2, buf1, H_DIM);

    // out = hidden + sigmoid(g2) * reasoned
    final_gate_kernel<<<(unsigned)(M * H_DIM / 256), 256, 0, stream>>>(
        hs, buf1, reasoned, out, M * H_DIM);

    (void)in_sizes; (void)n_in; (void)out_size; (void)ws_size;
}
